// LSTM_79568564126523
// MI455X (gfx1250) — matrix operand order repaired and checked by reading, not compiled, not run on hardware
//
#include <hip/hip_runtime.h>

#define VOCABN 8192
#define HIDN   1024
#define BATCHN 64
#define SEQN   256
#define NSTEP_BLOCKS 64   // persistent recurrent kernel grid size

typedef __attribute__((ext_vector_type(16))) __bf16 v16bf;
typedef __attribute__((ext_vector_type(8)))  float  v8f;

union ABTile { uint4 q[2]; v16bf v; };

__device__ __forceinline__ float sigm(float x) { return 1.0f / (1.0f + __expf(-x)); }

// ---------------- prep kernels (one-time, L2-resident afterwards) ----------------

// WtG[g][n][k] = bf16(W_h<g>[k][n])   (4 x 1024 x 1024)
__global__ __launch_bounds__(256) void prep_gates(
    const float* __restrict__ Wi, const float* __restrict__ Wf,
    const float* __restrict__ Wo, const float* __restrict__ Wc,
    __bf16* __restrict__ WtG) {
  size_t idx = (size_t)blockIdx.x * 256 + threadIdx.x;      // over 4*HID*HID
  size_t per = (size_t)HIDN * HIDN;
  int g = (int)(idx / per);
  size_t rem = idx % per;
  int n = (int)(rem / HIDN);
  int k = (int)(rem % HIDN);
  const float* W = (g == 0) ? Wi : (g == 1) ? Wf : (g == 2) ? Wo : Wc;
  WtG[idx] = (__bf16)W[(size_t)k * HIDN + n];
}

// WhyT[n][k] = bf16(W_hy[k][n])   (8192 x 1024)
__global__ __launch_bounds__(256) void prep_why(
    const float* __restrict__ Why, __bf16* __restrict__ WhyT) {
  size_t idx = (size_t)blockIdx.x * 256 + threadIdx.x;      // over VOCAB*HID
  int n = (int)(idx / HIDN);
  int k = (int)(idx % HIDN);
  WhyT[idx] = (__bf16)Why[(size_t)k * VOCABN + n];
}

// Hall slot 0 = bf16(H0); also zero the grid-barrier counter (determinism per call)
__global__ __launch_bounds__(256) void prep_h0(
    const float* __restrict__ H0, __bf16* __restrict__ Hall0,
    unsigned* __restrict__ barrier_cnt) {
  size_t idx = (size_t)blockIdx.x * 256 + threadIdx.x;      // over BATCH*HID
  Hall0[idx] = (__bf16)H0[idx];
  if (idx == 0) *barrier_cnt = 0u;
}

// ---------------- persistent recurrent kernel: all SEQ steps in one launch ----------
// grid 64 (N tiles of 16), block 128 (4 waves = 4 M tiles of 16).
// Device-scope atomic grid barrier between timesteps (64 blocks << WGP count,
// guaranteed co-resident). Weights stay hot in L2 across all 256 steps.
__global__ __launch_bounds__(128) void lstm_seq(
    const int*   __restrict__ x,
    const float* __restrict__ Wxi, const float* __restrict__ bi,
    const float* __restrict__ Wxf, const float* __restrict__ bfg,
    const float* __restrict__ Wxo, const float* __restrict__ bo,
    const float* __restrict__ Wxc, const float* __restrict__ bc,
    const __bf16* __restrict__ WtG,   // [4][HID][HID]  (n-major, k contiguous)
    __bf16* __restrict__ Hall,        // [(SEQ+1)][BATCH][HID]
    unsigned* __restrict__ barrier_cnt,
    float* __restrict__ finalH, float* __restrict__ finalC) {
  const int lane = threadIdx.x & 31;
  const int wv   = threadIdx.x >> 5;   // M tile 0..3
  const int hlf  = lane >> 4;
  const int l16  = lane & 15;
  const int ncol = blockIdx.x * 16 + l16;

  const float* Wx[4]   = {Wxi, Wxf, Wxo, Wxc};
  float biasv[4] = {bi[ncol], bfg[ncol], bo[ncol], bc[ncol]};

  int rows[8];
#pragma unroll
  for (int r = 0; r < 8; ++r) rows[r] = wv * 16 + hlf * 8 + r;   // C/D layout rows

  for (int t = 0; t < SEQN; ++t) {
    const __bf16* Hprev = Hall + (size_t)t * (BATCHN * HIDN);

    // acc init: one_hot(x)@W_x == W_x[x]  (fp32 gather) + bias
    v8f acc[4];
#pragma unroll
    for (int g = 0; g < 4; ++g) {
#pragma unroll
      for (int r = 0; r < 8; ++r) {
        int tok = x[rows[r] * SEQN + t];
        acc[g][r] = Wx[g][(size_t)tok * HIDN + ncol] + biasv[g];
      }
    }

    const __bf16* aRow = Hprev + (size_t)(wv * 16 + l16) * HIDN;
    for (int kk = 0; kk < HIDN / 32; ++kk) {
      ABTile A;
      const __bf16* ap = aRow + kk * 32 + hlf * 8;          // K = hlf*8+{0..7}
      A.q[0] = *(const uint4*)(ap);
      A.q[1] = *(const uint4*)(ap + 16);                    // K = 16+hlf*8+{0..7}
#pragma unroll
      for (int g = 0; g < 4; ++g) {
        ABTile B;
        const __bf16* bp = WtG + ((size_t)g * HIDN + ncol) * HIDN + kk * 32 + hlf * 8;
        B.q[0] = *(const uint4*)(bp);                       // K = hlf*8+{0..7}
        B.q[1] = *(const uint4*)(bp + 16);                  // K = 16+hlf*8+{0..7}
        acc[g] = __builtin_amdgcn_wmma_f32_16x16x32_bf16(
            false, A.v, false, B.v, (short)0, acc[g], false, false);
      }
    }

    __bf16* Hnext = Hall + (size_t)(t + 1) * (BATCHN * HIDN);
#pragma unroll
    for (int r = 0; r < 8; ++r) {
      float ig = sigm(acc[0][r]);
      float fg = sigm(acc[1][r]);
      float og = sigm(acc[2][r]);
      float ct = tanhf(acc[3][r]);
      float c  = ig * ct + fg * og;    // faithful to (buggy) reference
      float h  = og * tanhf(c);
      size_t off = (size_t)rows[r] * HIDN + ncol;
      Hnext[off] = (__bf16)h;
      if (t == SEQN - 1) { finalH[off] = h; finalC[off] = c; }
    }

    // ---- grid-wide barrier: release our H stores, acquire everyone else's ----
    __threadfence();                   // flush Hnext stores to device scope
    __syncthreads();
    if (threadIdx.x == 0) {
      __hip_atomic_fetch_add(barrier_cnt, 1u, __ATOMIC_ACQ_REL,
                             __HIP_MEMORY_SCOPE_AGENT);
      unsigned target = (unsigned)NSTEP_BLOCKS * (unsigned)(t + 1);
      while (__hip_atomic_load(barrier_cnt, __ATOMIC_ACQUIRE,
                               __HIP_MEMORY_SCOPE_AGENT) < target) {
        __builtin_amdgcn_s_sleep(1);
      }
    }
    __syncthreads();
    __threadfence();                   // invalidate stale lines before next read
  }
}

// ---------------- big deferred output GEMM: Y = Hall[1..S] @ W_hy + b_y ----------------
// grid (VOCAB/128, S*B/128), block 256 (8 waves): wave = (mq 0..3, nq 0..1).
// Each wave: 32M x 64N strip = 2x4 accumulators (A reused x4, B reused x2),
// halving L2 traffic vs 1x4 blocking. Nontemporal fp32 output stores.
__global__ __launch_bounds__(256) void out_gemm(
    const __bf16* __restrict__ H1,    // [S*B][HID] bf16
    const __bf16* __restrict__ WhyT,  // [VOCAB][HID] bf16
    const float*  __restrict__ by,
    float* __restrict__ out) {
  const int lane  = threadIdx.x & 31;
  const int wv    = threadIdx.x >> 5;
  const int hlf   = lane >> 4;
  const int l16   = lane & 15;
  const int mq    = wv >> 1;                       // 0..3
  const int nq    = wv & 1;                        // 0..1
  const int mbase = blockIdx.y * 128 + mq * 32;    // 2 M-tiles per wave
  const int nbase = blockIdx.x * 128 + nq * 64;    // 4 N-tiles per wave

  v8f acc[2][4];
#pragma unroll
  for (int j = 0; j < 4; ++j) {
    float b = by[nbase + j * 16 + l16];
    v8f s = (v8f){b, b, b, b, b, b, b, b};
    acc[0][j] = s;
    acc[1][j] = s;
  }

  const __bf16* aRow0 = H1 + (size_t)(mbase + l16) * HIDN;
  const __bf16* aRow1 = H1 + (size_t)(mbase + 16 + l16) * HIDN;
  for (int kk = 0; kk < HIDN / 32; ++kk) {
    ABTile A0, A1;
    {
      const __bf16* ap0 = aRow0 + kk * 32 + hlf * 8;
      A0.q[0] = *(const uint4*)(ap0);
      A0.q[1] = *(const uint4*)(ap0 + 16);
      const __bf16* ap1 = aRow1 + kk * 32 + hlf * 8;
      A1.q[0] = *(const uint4*)(ap1);
      A1.q[1] = *(const uint4*)(ap1 + 16);
    }
#pragma unroll
    for (int j = 0; j < 4; ++j) {
      ABTile B;
      const __bf16* bp = WhyT + (size_t)(nbase + j * 16 + l16) * HIDN + kk * 32 + hlf * 8;
      B.q[0] = *(const uint4*)(bp);
      B.q[1] = *(const uint4*)(bp + 16);
      acc[0][j] = __builtin_amdgcn_wmma_f32_16x16x32_bf16(
          false, A0.v, false, B.v, (short)0, acc[0][j], false, false);
      acc[1][j] = __builtin_amdgcn_wmma_f32_16x16x32_bf16(
          false, A1.v, false, B.v, (short)0, acc[1][j], false, false);
    }
  }

#pragma unroll
  for (int mi = 0; mi < 2; ++mi) {
#pragma unroll
    for (int j = 0; j < 4; ++j) {
#pragma unroll
      for (int r = 0; r < 8; ++r) {
        int mrow = mbase + mi * 16 + hlf * 8 + r;
        int ncol = nbase + j * 16 + l16;
        __builtin_nontemporal_store(acc[mi][j][r], out + (size_t)mrow * VOCABN + ncol);
      }
    }
  }
}

// ---------------- host side ----------------
extern "C" void kernel_launch(void* const* d_in, const int* in_sizes, int n_in,
                              void* d_out, int out_size, void* d_ws, size_t ws_size,
                              hipStream_t stream) {
  (void)in_sizes; (void)n_in; (void)out_size; (void)ws_size;
  const int*   x   = (const int*)  d_in[0];
  const float* H0  = (const float*)d_in[1];
  // d_in[2] (C0) is never read by the (bug-faithful) recurrence.
  const float* Wxi = (const float*)d_in[3];
  const float* Whi = (const float*)d_in[4];
  const float* bi  = (const float*)d_in[5];
  const float* Wxf = (const float*)d_in[6];
  const float* Whf = (const float*)d_in[7];
  const float* bfg = (const float*)d_in[8];
  const float* Wxo = (const float*)d_in[9];
  const float* Who = (const float*)d_in[10];
  const float* bo  = (const float*)d_in[11];
  const float* Wxc = (const float*)d_in[12];
  const float* Whc = (const float*)d_in[13];
  const float* bc  = (const float*)d_in[14];
  const float* Why = (const float*)d_in[15];
  const float* by  = (const float*)d_in[16];

  // workspace carve (all offsets 256B-aligned)
  char* ws = (char*)d_ws;
  const size_t hallBytes = (size_t)(SEQN + 1) * BATCHN * HIDN * 2;  // 33,685,504
  const size_t wtgBytes  = (size_t)4 * HIDN * HIDN * 2;             //  8,388,608
  const size_t whyBytes  = (size_t)VOCABN * HIDN * 2;               // 16,777,216
  __bf16*   Hall = (__bf16*)ws;
  __bf16*   WtG  = (__bf16*)(ws + hallBytes);
  __bf16*   WhyT = (__bf16*)(ws + hallBytes + wtgBytes);
  unsigned* bcnt = (unsigned*)(ws + hallBytes + wtgBytes + whyBytes);

  float* out    = (float*)d_out;
  float* finalH = out + (size_t)SEQN * BATCHN * VOCABN;
  float* finalC = finalH + (size_t)BATCHN * HIDN;

  prep_gates<<<(4u * HIDN * HIDN) / 256, 256, 0, stream>>>(Whi, Whf, Who, Whc, WtG);
  prep_why<<<((unsigned)VOCABN * HIDN) / 256, 256, 0, stream>>>(Why, WhyT);
  prep_h0<<<((unsigned)BATCHN * HIDN) / 256, 256, 0, stream>>>(H0, Hall, bcnt);

  lstm_seq<<<NSTEP_BLOCKS, 128, 0, stream>>>(
      x, Wxi, bi, Wxf, bfg, Wxo, bo, Wxc, bc, WtG, Hall, bcnt, finalH, finalC);

  out_gemm<<<dim3(VOCABN / 128, (SEQN * BATCHN) / 128), 256, 0, stream>>>(
      Hall + (size_t)BATCHN * HIDN, WhyT, by, out);
}
